// TriangleAttention_15564961481077
// MI455X (gfx1250) — matrix operand order repaired and checked by reading, not compiled, not run on hardware
//
#include <hip/hip_runtime.h>

// ---------------------------------------------------------------- types ----
typedef __attribute__((ext_vector_type(16))) __bf16 v16bf;
typedef __attribute__((ext_vector_type(8)))  float  v8f;
typedef __attribute__((ext_vector_type(4)))  unsigned int u32x4;
typedef __attribute__((ext_vector_type(8)))  int  i32x8;
typedef __attribute__((ext_vector_type(4)))  int  i32x4;

#define L_  320
#define DP_ 128
#define NH_ 4
#define DH_ 32
#define LL_ (L_*L_)

__device__ __forceinline__ unsigned short f2bf(float f) {
  unsigned int u = __float_as_uint(f);
  u += 0x7FFFu + ((u >> 16) & 1u);            // round-to-nearest-even
  return (unsigned short)(u >> 16);
}
__device__ __forceinline__ float bf2f(unsigned short s) {
  return __uint_as_float(((unsigned int)s) << 16);
}

struct V16B { union { v16bf v; uint4 u[2]; }; };

__device__ __forceinline__ v16bf ld16(const unsigned short* p0,
                                      const unsigned short* p1) {
  V16B t;
  t.u[0] = *(const uint4*)p0;   // 16B aligned by construction
  t.u[1] = *(const uint4*)p1;
  return t.v;
}

__device__ __forceinline__ v8f wmma_bf16(v16bf a, v16bf b, v8f c) {
  return __builtin_amdgcn_wmma_f32_16x16x32_bf16(
      false, a, false, b, (short)0, c, false, false);
}

// ---- Tensor Data Mover: DMA one 128x128 bf16 matrix (32KB) global -> LDS --
// D# per CDNA5 ISA (2D tile == whole tensor).
// This toolchain exposes the 6-arg builtin: (g0, g1, g2, g3, g_extra, cpol).
__device__ __forceinline__ void tdm_load_w128(const unsigned short* gsrc,
                                              unsigned short* lds_dst) {
  unsigned long long ga = (unsigned long long)(uintptr_t)gsrc;
  unsigned int laddr = (unsigned int)(uintptr_t)lds_dst;   // LDS byte offset
  u32x4 g0 = { 1u,                                  // count=1 valid descriptor
               laddr,                               // lds_addr [63:32]
               (unsigned int)(ga & 0xFFFFFFFFull),  // global_addr lo
               (unsigned int)((ga >> 32) & 0x1FFFFFFull) | (2u << 30) }; // type=2
  i32x8 g1 = { (int)0x10000,          // data_size=1 (2B), mask=0
               (int)(128u << 16),     // tensor_dim0 = 128   (bits 79:48 lo)
               (int)(128u << 16),     // tensor_dim1 = 128   (bits 111:80 lo)
               (int)(128u << 16),     // tile_dim0   = 128   (bits 127:112)
               (int)128,              // tile_dim1   = 128   (bits 143:128)
               (int)128,              // tensor_dim0_stride = 128 (lo32)
               (int)(16384u << 16),   // tensor_dim1_stride = 16384 (lo16)
               0 };
  i32x4 gz4 = { 0, 0, 0, 0 };
  i32x8 gz8 = { 0, 0, 0, 0, 0, 0, 0, 0 };
  __builtin_amdgcn_tensor_load_to_lds(g0, g1, gz4, gz4, gz8, 0);
}

// ------------------------------------------------ weight fp32 -> bf16 ------
__global__ __launch_bounds__(256) void k_cvtw(const float* Wq, const float* Wk,
                                              const float* Wv, const float* Wg,
                                              const float* Wo,
                                              unsigned short* dst) {
  int idx = blockIdx.x * 256 + threadIdx.x;           // 5*16384 = 81920
  if (idx >= 5 * DP_ * DP_) return;
  int m = idx >> 14, off = idx & 16383;
  const float* src = (m == 0) ? Wq : (m == 1) ? Wk : (m == 2) ? Wv
                   : (m == 3) ? Wg : Wo;
  dst[idx] = f2bf(src[off]);
}

// -------------------------- fused LayerNorm (-> x bf16) + pair@Wb^T bias ---
__global__ __launch_bounds__(256) void k_prep(const float* __restrict__ pair,
                                              const float* __restrict__ nw,
                                              const float* __restrict__ nb,
                                              const float* __restrict__ Wb,
                                              unsigned short* __restrict__ xbf,
                                              float* __restrict__ bias) {
  int wave = threadIdx.x >> 5, lane = threadIdx.x & 31;
  int p = blockIdx.x * 8 + wave;                      // (i,j) flat, < 102400
  const float4 v = ((const float4*)(pair + (long)p * DP_))[lane];

  float s  = v.x + v.y + v.z + v.w;
  float sq = v.x*v.x + v.y*v.y + v.z*v.z + v.w*v.w;
  #pragma unroll
  for (int off = 16; off >= 1; off >>= 1) {
    s  += __shfl_xor(s,  off, 32);
    sq += __shfl_xor(sq, off, 32);
  }
  float mean = s * (1.0f / 128.0f);
  float var  = sq * (1.0f / 128.0f) - mean * mean;
  float rstd = rsqrtf(var + 1e-5f);

  float4 w4 = ((const float4*)nw)[lane];
  float4 b4 = ((const float4*)nb)[lane];
  unsigned int o0 = f2bf((v.x - mean) * rstd * w4.x + b4.x);
  unsigned int o1 = f2bf((v.y - mean) * rstd * w4.y + b4.y);
  unsigned int o2 = f2bf((v.z - mean) * rstd * w4.z + b4.z);
  unsigned int o3 = f2bf((v.w - mean) * rstd * w4.w + b4.w);
  uint2 pk; pk.x = o0 | (o1 << 16); pk.y = o2 | (o3 << 16);
  *(uint2*)(xbf + (long)p * DP_ + lane * 4) = pk;

  // bias layout ((i*4+h)*320 + j), from RAW pair
  int i = p / L_, j = p - i * L_;
  #pragma unroll
  for (int h = 0; h < NH_; ++h) {
    float4 wb4 = ((const float4*)(Wb + h * DP_))[lane];
    float d = v.x*wb4.x + v.y*wb4.y + v.z*wb4.z + v.w*wb4.w;
    #pragma unroll
    for (int off = 16; off >= 1; off >>= 1) d += __shfl_xor(d, off, 32);
    if (lane == h) bias[(i * NH_ + h) * L_ + j] = d;
  }
}

// ---- fused Q/K/V/G projection: A read once, weights TDM double-buffered ---
// mat 0: Q head layout bf16   ((i*4+h)*320 + j)*32 + dh
// mat 1: K head layout bf16
// mat 2: V transposed bf16    ((i*4+h)*32 + dh)*320 + j
// mat 3: sigmoid(acc+bg) bf16 head layout (gate)
__global__ __launch_bounds__(256) void k_proj4(const unsigned short* __restrict__ A,
                                               const unsigned short* __restrict__ Wqkvg,
                                               const float* __restrict__ bg,
                                               unsigned short* __restrict__ Qh,
                                               unsigned short* __restrict__ Kh,
                                               unsigned short* __restrict__ Vt,
                                               unsigned short* __restrict__ Gh) {
  __shared__ unsigned short wsh[2][DP_ * DP_];        // 2 x 32KB
  int wave = threadIdx.x >> 5, lane = threadIdx.x & 31;
  int half = lane >> 4, ln = lane & 15;
  int R0 = blockIdx.x * 128 + wave * 16;              // this wave's 16 rows
  int grow0 = R0 + half * 8;

  // A tiles for all 4 k-steps, loaded once (one clause of 8 x b128)
  v16bf a[4];
  const unsigned short* arow = A + (long)(R0 + ln) * DP_;
  #pragma unroll
  for (int kt = 0; kt < 4; ++kt) {
    int ka = kt * 32 + half * 16;
    a[kt] = ld16(arow + ka, arow + ka + 8);
  }

  // stage W0, then per-iteration overlap TDM(m+1) with compute(m)
  if (threadIdx.x == 0) tdm_load_w128(Wqkvg, &wsh[0][0]);
  __builtin_amdgcn_s_wait_tensorcnt(0);
  __syncthreads();

  #pragma unroll
  for (int mat = 0; mat < 4; ++mat) {
    if (threadIdx.x == 0 && mat < 3)
      tdm_load_w128(Wqkvg + (mat + 1) * DP_ * DP_, &wsh[(mat + 1) & 1][0]);

    const unsigned short* wbase = &wsh[mat & 1][0];
    v8f acc[8];
    #pragma unroll
    for (int ct = 0; ct < 8; ++ct) acc[ct] = (v8f){0,0,0,0,0,0,0,0};

    #pragma unroll
    for (int kt = 0; kt < 4; ++kt) {
      v16bf b[8];
      #pragma unroll
      for (int ct = 0; ct < 8; ++ct) {
        const unsigned short* wrow =
            wbase + (ct * 16 + ln) * DP_ + kt * 32 + half * 16;
        b[ct] = ld16(wrow, wrow + 8);
      }
      #pragma unroll
      for (int ct = 0; ct < 8; ++ct) acc[ct] = wmma_bf16(a[kt], b[ct], acc[ct]);
    }

    // epilogue
    #pragma unroll
    for (int ct = 0; ct < 8; ++ct) {
      int c = ct * 16 + ln, hh = c >> 5, dh = c & 31;
      #pragma unroll
      for (int r = 0; r < 8; ++r) {
        int grow = grow0 + r;
        int i = grow / L_, j = grow - i * L_;
        float val = acc[ct][r];
        if (mat == 0) {
          Qh[((long)(i * NH_ + hh) * L_ + j) * DH_ + dh] = f2bf(val);
        } else if (mat == 1) {
          Kh[((long)(i * NH_ + hh) * L_ + j) * DH_ + dh] = f2bf(val);
        } else if (mat == 2) {
          Vt[((long)((i * NH_ + hh) * DH_ + dh)) * L_ + j] = f2bf(val);
        } else {
          float g = 1.0f / (1.0f + __expf(-(val + bg[c])));
          Gh[((long)(i * NH_ + hh) * L_ + j) * DH_ + dh] = f2bf(g);
        }
      }
    }

    __builtin_amdgcn_s_wait_tensorcnt(0);   // next buffer resident (wave 0)
    __syncthreads();                        // all waves done with old buffer
  }
}

// --------------------------- attention: one (i,h) per block, 20 waves ------
__global__ __launch_bounds__(640, 1) void k_attn(const unsigned short* __restrict__ Qh,
                                                 const unsigned short* __restrict__ Kh,
                                                 const unsigned short* __restrict__ Vt,
                                                 const unsigned short* __restrict__ Gh,
                                                 const float* __restrict__ bias,
                                                 unsigned short* __restrict__ gated) {
  __shared__ unsigned short pslab[20 * 512];          // 16x32 bf16 per wave
  int i = blockIdx.x, h = blockIdx.y;
  int wave = threadIdx.x >> 5, lane = threadIdx.x & 31;
  int half = lane >> 4, ln = lane & 15;
  int rt = wave;                                      // row tile 0..19

  const unsigned short* Qb = Qh + (long)(i * NH_ + h) * L_ * DH_;  // [j][dh]
  const unsigned short* Kb = Kh + (long)(i * NH_ + h) * L_ * DH_;  // [j][dh]
  const unsigned short* Vb = Vt + (long)(i * NH_ + h) * DH_ * L_;  // [dh][j]
  const float* brow = bias + (i * NH_ + h) * L_;

  const unsigned short* qp = Qb + (rt * 16 + ln) * DH_ + half * 16;
  v16bf aQ = ld16(qp, qp + 8);

  // ---- S = Q K^T : 20 column tiles, loads batched 5 at a time ----
  v8f acc[20];
  #pragma unroll
  for (int g = 0; g < 4; ++g) {
    v16bf bb[5];
    #pragma unroll
    for (int t = 0; t < 5; ++t) {
      const unsigned short* kp = Kb + ((g * 5 + t) * 16 + ln) * DH_ + half * 16;
      bb[t] = ld16(kp, kp + 8);
    }
    #pragma unroll
    for (int t = 0; t < 5; ++t)
      acc[g * 5 + t] = wmma_bf16(aQ, bb[t], (v8f){0,0,0,0,0,0,0,0});
  }

  // ---- scale + bias, row max/sum (rows live in 16-lane halves) ----
  const float scale = 0.17677669529663687f;           // 1/sqrt(32)
  float mx[8], sm[8], rinv[8];
  #pragma unroll
  for (int r = 0; r < 8; ++r) mx[r] = -3.0e38f;
  #pragma unroll
  for (int nt = 0; nt < 20; ++nt) {
    float bv = brow[nt * 16 + ln];
    #pragma unroll
    for (int r = 0; r < 8; ++r) {
      acc[nt][r] = acc[nt][r] * scale + bv;
      mx[r] = fmaxf(mx[r], acc[nt][r]);
    }
  }
  #pragma unroll
  for (int r = 0; r < 8; ++r) {
    #pragma unroll
    for (int off = 8; off >= 1; off >>= 1)
      mx[r] = fmaxf(mx[r], __shfl_xor(mx[r], off, 32));
    sm[r] = 0.0f;
  }
  #pragma unroll
  for (int nt = 0; nt < 20; ++nt)
    #pragma unroll
    for (int r = 0; r < 8; ++r) {
      float p = __expf(acc[nt][r] - mx[r]);
      acc[nt][r] = p;
      sm[r] += p;
    }
  #pragma unroll
  for (int r = 0; r < 8; ++r) {
    #pragma unroll
    for (int off = 8; off >= 1; off >>= 1) sm[r] += __shfl_xor(sm[r], off, 32);
    rinv[r] = 1.0f / sm[r];
  }

  // ---- O = P V : stage P (D-layout) -> bf16 A-layout through private LDS
  unsigned short* slab = pslab + wave * 512;
  v8f oacc[2];
  oacc[0] = (v8f){0,0,0,0,0,0,0,0};
  oacc[1] = (v8f){0,0,0,0,0,0,0,0};
  #pragma unroll
  for (int kt = 0; kt < 10; ++kt) {
    #pragma unroll
    for (int r = 0; r < 8; ++r) {
      int m = r + half * 8;
      slab[m * 32 + ln]      = f2bf(acc[2 * kt][r]);
      slab[m * 32 + 16 + ln] = f2bf(acc[2 * kt + 1][r]);
    }
    const unsigned short* pp = slab + ln * 32 + half * 16;
    v16bf aP = ld16(pp, pp + 8);
    const unsigned short* vp0 = Vb + ln * L_ + kt * 32 + half * 16;
    v16bf b0 = ld16(vp0, vp0 + 8);
    const unsigned short* vp1 = Vb + (16 + ln) * L_ + kt * 32 + half * 16;
    v16bf b1 = ld16(vp1, vp1 + 8);
    oacc[0] = wmma_bf16(aP, b0, oacc[0]);
    oacc[1] = wmma_bf16(aP, b1, oacc[1]);
  }

  // ---- epilogue: 1/rowsum, gate, store gated bf16 row-major (LL x 128) ----
  #pragma unroll
  for (int ntv = 0; ntv < 2; ++ntv)
    #pragma unroll
    for (int r = 0; r < 8; ++r) {
      int mg = rt * 16 + r + half * 8;                // query row in [0,320)
      float g = bf2f(Gh[((long)(i * NH_ + h) * L_ + mg) * DH_ + ntv * 16 + ln]);
      float val = oacc[ntv][r] * rinv[r] * g;
      gated[((long)i * L_ + mg) * DP_ + h * DH_ + ntv * 16 + ln] = f2bf(val);
    }
}

// ---------------- output projection: out = gated@Wo^T + bo + pair ----------
__global__ __launch_bounds__(256) void k_outproj(const unsigned short* __restrict__ A,
                                                 const unsigned short* __restrict__ Wo,
                                                 const float* __restrict__ bo,
                                                 const float* __restrict__ pair,
                                                 float* __restrict__ outf) {
  __shared__ unsigned short wsh[DP_ * DP_];           // 32KB
  int wave = threadIdx.x >> 5, lane = threadIdx.x & 31;
  int half = lane >> 4, ln = lane & 15;
  int R0 = blockIdx.x * 128 + wave * 16;

  v16bf a[4];
  const unsigned short* arow = A + (long)(R0 + ln) * DP_;
  #pragma unroll
  for (int kt = 0; kt < 4; ++kt) {
    int ka = kt * 32 + half * 16;
    a[kt] = ld16(arow + ka, arow + ka + 8);
  }

  if (threadIdx.x == 0) tdm_load_w128(Wo, &wsh[0]);
  __builtin_amdgcn_s_wait_tensorcnt(0);
  __syncthreads();

  v8f acc[8];
  #pragma unroll
  for (int ct = 0; ct < 8; ++ct) acc[ct] = (v8f){0,0,0,0,0,0,0,0};
  #pragma unroll
  for (int kt = 0; kt < 4; ++kt) {
    v16bf b[8];
    #pragma unroll
    for (int ct = 0; ct < 8; ++ct) {
      const unsigned short* wrow = wsh + (ct * 16 + ln) * DP_ + kt * 32 + half * 16;
      b[ct] = ld16(wrow, wrow + 8);
    }
    #pragma unroll
    for (int ct = 0; ct < 8; ++ct) acc[ct] = wmma_bf16(a[kt], b[ct], acc[ct]);
  }

  #pragma unroll
  for (int ct = 0; ct < 8; ++ct) {
    int c = ct * 16 + ln;
    #pragma unroll
    for (int r = 0; r < 8; ++r) {
      int grow = R0 + r + half * 8;
      outf[(long)grow * DP_ + c] = acc[ct][r] + bo[c] + pair[(long)grow * DP_ + c];
    }
  }
}

// ------------------------------------------------------------- launcher ----
extern "C" void kernel_launch(void* const* d_in, const int* in_sizes, int n_in,
                              void* d_out, int out_size, void* d_ws, size_t ws_size,
                              hipStream_t stream) {
  (void)in_sizes; (void)n_in; (void)out_size; (void)ws_size;
  const float* pair   = (const float*)d_in[0];
  const float* norm_w = (const float*)d_in[1];
  const float* norm_b = (const float*)d_in[2];
  const float* Wq     = (const float*)d_in[3];
  const float* Wk     = (const float*)d_in[4];
  const float* Wv     = (const float*)d_in[5];
  const float* Wg     = (const float*)d_in[6];
  const float* bg     = (const float*)d_in[7];
  const float* Wo     = (const float*)d_in[8];
  const float* bo     = (const float*)d_in[9];
  const float* Wb     = (const float*)d_in[10];

  char* ws = (char*)d_ws;
  const size_t SZ_BF = (size_t)LL_ * DP_ * 2;         // 26,214,400 B
  unsigned short* xbf   = (unsigned short*)(ws);
  unsigned short* Qh    = (unsigned short*)(ws + 1 * SZ_BF);
  unsigned short* Kh    = (unsigned short*)(ws + 2 * SZ_BF);
  unsigned short* Vt    = (unsigned short*)(ws + 3 * SZ_BF);
  unsigned short* Gh    = (unsigned short*)(ws + 4 * SZ_BF);
  unsigned short* gated = (unsigned short*)(ws + 5 * SZ_BF);
  float*          bias  = (float*)(ws + 6 * SZ_BF);   // L*H*L fp32
  unsigned short* Wbf   = (unsigned short*)(ws + 6 * SZ_BF + (size_t)L_ * NH_ * L_ * 4);
  unsigned short* Wob   = Wbf + 4 * DP_ * DP_;        // Wq..Wg then Wo

  k_cvtw<<<320, 256, 0, stream>>>(Wq, Wk, Wv, Wg, Wo, Wbf);
  k_prep<<<LL_ / 8, 256, 0, stream>>>(pair, norm_w, norm_b, Wb, xbf, bias);

  k_proj4<<<LL_ / 128, 256, 0, stream>>>(xbf, Wbf, bg, Qh, Kh, Vt, Gh);

  k_attn<<<dim3(L_, NH_), 640, 0, stream>>>(Qh, Kh, Vt, Gh, bias, gated);

  k_outproj<<<LL_ / 128, 256, 0, stream>>>(gated, Wob, bo, pair, (float*)d_out);
}
